// SimpleMambaBlock_32916629357355
// MI455X (gfx1250) — matrix operand order repaired and checked by reading, not compiled, not run on hardware
//
#include <hip/hip_runtime.h>
#include <math.h>

#define DM     1024
#define DI     2048
#define NXZ    4096
#define DSTATE 16
#define DCONV  4
#define BB     4
#define LL     2048
#define MTOK   (BB*LL)   /* 8192 tokens */

typedef __attribute__((ext_vector_type(16))) __bf16       v16bf;
typedef __attribute__((ext_vector_type(8)))  float        v8f;
typedef __attribute__((ext_vector_type(4)))  unsigned int v4u;
typedef __attribute__((ext_vector_type(8)))  unsigned int v8u;
typedef __attribute__((ext_vector_type(4)))  unsigned int u32x4;
typedef __attribute__((ext_vector_type(4)))  int          i32x4;
typedef __attribute__((ext_vector_type(8)))  int          i32x8;

__device__ __forceinline__ unsigned short f2bf(float f) {
  unsigned int u = __float_as_uint(f);
  u += 0x7FFFu + ((u >> 16) & 1u);          // round-to-nearest-even
  return (unsigned short)(u >> 16);
}

__device__ __forceinline__ v16bf make_frag(v4u lo, v4u hi) {
  v8u c;
  c[0]=lo[0]; c[1]=lo[1]; c[2]=lo[2]; c[3]=lo[3];
  c[4]=hi[0]; c[5]=hi[1]; c[6]=hi[2]; c[7]=hi[3];
  return __builtin_bit_cast(v16bf, c);
}

/* ---- TDM: DMA a 2-D bf16 tile (tileRows x tileK) from global into LDS ----
 * D# packed per CDNA5 ISA sec 8.3-8.6: group0 = {count, lds_addr, global_addr, type=2},
 * group1 = {data_size=2B, tensor_dim0/1, tile_dim0/1, tensor_dim0_stride}. Not in a
 * cluster -> workgroup_mask = 0. Groups 2/3 zero (2-D tile, tile_dim2 = 0 = unused). */
__device__ __forceinline__ void tdm_load_tile_bf16(
    unsigned ldsOff, const void* gptr,
    unsigned tileK, unsigned tileRows, unsigned long long rowStrideElems)
{
  unsigned long long ga = (unsigned long long)(uintptr_t)gptr;
  u32x4 g0;
  g0[0] = 1u;                                         /* count=1, user descriptor  */
  g0[1] = ldsOff;                                     /* lds_addr (bytes)          */
  g0[2] = (unsigned)(ga & 0xFFFFFFFFu);               /* global_addr[31:0]         */
  g0[3] = (unsigned)((ga >> 32) & 0x01FFFFFFu)        /* global_addr[56:32]        */
        | 0x80000000u;                                /* type=2 ("image")          */
  i32x8 g1;
  g1[0] = (int)(1u << 16);                            /* data_size=1 -> 2 bytes    */
  g1[1] = (int)((tileK & 0xFFFFu) << 16);             /* tensor_dim0 lo16 @63:48   */
  g1[2] = (int)((tileK >> 16) | ((tileRows & 0xFFFFu) << 16)); /* tdim0 hi|tdim1 lo*/
  g1[3] = (int)((tileRows >> 16) | ((tileK & 0xFFFFu) << 16)); /* tdim1 hi|tile_dim0*/
  g1[4] = (int)(tileRows & 0xFFFFu);                  /* tile_dim1 | tile_dim2=0   */
  g1[5] = (int)(rowStrideElems & 0xFFFFFFFFull);      /* tensor_dim0_stride lo32   */
  g1[6] = (int)((rowStrideElems >> 32) & 0xFFFFull);  /* stride0 hi16 | stride1=0  */
  g1[7] = 0;
  i32x4 z4 = {0, 0, 0, 0};
#if defined(__clang_major__) && (__clang_major__ >= 23)
  i32x8 z8 = {0, 0, 0, 0, 0, 0, 0, 0};
  __builtin_amdgcn_tensor_load_to_lds(g0, g1, z4, z4, z8, 0);
#else
  __builtin_amdgcn_tensor_load_to_lds(g0, g1, z4, z4, 0);
#endif
}

/* ---------------- prep kernels ---------------- */

__global__ void k_cvt_bf16(const float* __restrict__ src,
                           unsigned short* __restrict__ dst, int n) {
  int i = blockIdx.x*blockDim.x + threadIdx.x;
  if (i < n) dst[i] = f2bf(src[i]);
}

__global__ void k_transpose_bf16(const float* __restrict__ src,
                                 unsigned short* __restrict__ dst,
                                 int rows, int cols) {
  int i = blockIdx.x*blockDim.x + threadIdx.x;
  if (i < rows*cols) {
    int r = i / cols, c = i % cols;
    dst[(size_t)c*rows + r] = f2bf(src[i]);
  }
}

__global__ void k_transpose_f32(const float* __restrict__ src,
                                float* __restrict__ dst, int rows, int cols) {
  int i = blockIdx.x*blockDim.x + threadIdx.x;
  if (i < rows*cols) {
    int r = i / cols, c = i % cols;
    dst[(size_t)c*rows + r] = src[i];
  }
}

__global__ void k_negexp(const float* __restrict__ a, float* __restrict__ o, int n) {
  int i = blockIdx.x*blockDim.x + threadIdx.x;
  if (i < n) o[i] = -__expf(a[i]);
}

/* ------------- TDM-staged double-buffered bf16 WMMA GEMM -------------------
 * C(MxN,f32) = A(MxK,bf16) * B, with B supplied as BT(NxK,bf16).
 * Block: 8 waves, 128x128 tile. Wave: 32x64 via 2x4 v_wmma_f32_16x16x32_bf16.
 * K staged in chunks of 64 through LDS by the Tensor Data Mover (wave 0),
 * double-buffered; sync via s_wait_tensorcnt + barriers.
 * Dynamic LDS: 2*(128*64 A + 128*64 B) bf16 = 64 KB. */

__global__ __launch_bounds__(256) void k_gemm_bf16(
    const unsigned short* __restrict__ A,
    const unsigned short* __restrict__ BT,
    float* __restrict__ C, int M, int N, int K)
{
  extern __shared__ unsigned short smemg[];
  unsigned short* sA = smemg;                 /* [2][128*64] */
  unsigned short* sB = smemg + 2*128*64;      /* [2][128*64] */

  const int lane = threadIdx.x & 31;
  const int w    = threadIdx.x >> 5;
  const int r    = lane & 15;
  const int half = lane >> 4;
  const int wm   = w & 3;          /* 4 wave rows of 32 */
  const int wn   = w >> 2;         /* 2 wave cols of 64 */
  const int mBlk = blockIdx.x * 128;
  const int nBlk = blockIdx.y * 128;

  v8f acc[2][4];
  #pragma unroll
  for (int i=0;i<2;++i)
    #pragma unroll
    for (int j=0;j<4;++j)
      #pragma unroll
      for (int e=0;e<8;++e) acc[i][j][e] = 0.f;

  const int nStage = K >> 6;                  /* K/64 */
  const unsigned ldsA0 = (unsigned)(uintptr_t)(void*)sA;
  const unsigned ldsB0 = (unsigned)(uintptr_t)(void*)sB;

  if (w == 0) {
    tdm_load_tile_bf16(ldsA0, A  + (size_t)mBlk*K, 64u, 128u, (unsigned long long)K);
    tdm_load_tile_bf16(ldsB0, BT + (size_t)nBlk*K, 64u, 128u, (unsigned long long)K);
  }

  for (int ks = 0; ks < nStage; ++ks) {
    if (w == 0) {
      if (ks + 1 < nStage) {
        const unsigned buf = (ks + 1) & 1;
        tdm_load_tile_bf16(ldsA0 + buf*16384u, A  + (size_t)mBlk*K + (ks+1)*64, 64u, 128u, (unsigned long long)K);
        tdm_load_tile_bf16(ldsB0 + buf*16384u, BT + (size_t)nBlk*K + (ks+1)*64, 64u, 128u, (unsigned long long)K);
        __builtin_amdgcn_s_wait_tensorcnt(2);  /* stage ks complete, ks+1 in flight */
      } else {
        __builtin_amdgcn_s_wait_tensorcnt(0);
      }
    }
    __syncthreads();

    const unsigned short* At = sA + (ks & 1)*16384;
    const unsigned short* Bt = sB + (ks & 1)*16384;
    #pragma unroll
    for (int kk = 0; kk < 64; kk += 32) {
      v16bf af[2];
      #pragma unroll
      for (int i=0;i<2;++i) {
        const unsigned short* p = At + (size_t)(wm*32 + i*16 + r)*64 + kk + 16*half;
        af[i] = make_frag(*(const v4u*)p, *(const v4u*)(p + 8));
      }
      v16bf bfr[4];
      #pragma unroll
      for (int j=0;j<4;++j) {
        const unsigned short* q = Bt + (size_t)(wn*64 + j*16 + r)*64 + kk + 16*half;
        bfr[j] = __builtin_bit_cast(v16bf, *(const v8u*)q);
      }
      #pragma unroll
      for (int i=0;i<2;++i)
        #pragma unroll
        for (int j=0;j<4;++j)
          acc[i][j] = __builtin_amdgcn_wmma_f32_16x16x32_bf16(
              false, af[i], false, bfr[j], (short)0, acc[i][j], false, false);
    }
    __syncthreads();   /* protect buffers before next TDM overwrite */
  }

  #pragma unroll
  for (int i=0;i<2;++i)
    #pragma unroll
    for (int j=0;j<4;++j) {
      float* cp = C + (size_t)(mBlk + wm*32 + i*16 + half*8) * N + (nBlk + wn*64 + j*16 + r);
      #pragma unroll
      for (int e=0;e<8;++e) cp[(size_t)e*N] = acc[i][j][e];
    }
}

/* ---------------- depthwise causal conv (width 4) + SiLU ---------------- */

__global__ void k_conv_silu(const float* __restrict__ xz,  /* M x NXZ, xc = cols [0,DI) */
                            const float* __restrict__ cw,  /* DI x DCONV */
                            const float* __restrict__ cb,
                            float* __restrict__ xs, int total) {
  int i = blockIdx.x*blockDim.x + threadIdx.x;
  if (i >= total) return;
  int d  = i % DI;
  int bl = i / DI;           /* b*L + l */
  int l  = bl % LL;
  float acc = cb[d];
  #pragma unroll
  for (int k = 0; k < DCONV; ++k) {
    int l2 = l + k - (DCONV - 1);
    if (l2 >= 0)
      acc = fmaf(xz[(size_t)(bl + (l2 - l))*NXZ + d], cw[d*DCONV + k], acc);
  }
  xs[i] = acc / (1.f + __expf(-acc));   /* silu */
}

/* ---------------- x_dbl = xs @ W_x  (33 tiny outputs per token, wave per token) ------- */

__global__ __launch_bounds__(128) void k_xdbl(
    const float* __restrict__ xs,
    const float* __restrict__ WxT,     /* 33 x DI (pre-transposed) */
    float* __restrict__ draw, float* __restrict__ Bp, float* __restrict__ Cp)
{
  __shared__ float sxs[4][DI];
  const int lane  = threadIdx.x & 31;
  const int wv    = threadIdx.x >> 5;
  const int token = blockIdx.x*4 + wv;
  for (int k = lane; k < DI; k += 32) sxs[wv][k] = xs[(size_t)token*DI + k];
  __syncthreads();
  for (int j = 0; j < 33; ++j) {
    float acc = 0.f;
    const float* wrow = WxT + (size_t)j*DI;
    for (int k = lane; k < DI; k += 32) acc = fmaf(sxs[wv][k], wrow[k], acc);
    #pragma unroll
    for (int off = 16; off; off >>= 1) acc += __shfl_xor(acc, off, 32);
    if (lane == 0) {
      if      (j == 0)       draw[token] = acc;
      else if (j <= DSTATE)  Bp[token*DSTATE + (j-1)]        = acc;
      else                   Cp[token*DSTATE + (j-1-DSTATE)] = acc;
    }
  }
}

/* ---------------- selective scan + gating, fused; chunked LDS staging ---------------- */

__global__ __launch_bounds__(256) void k_scan_gate(
    const float* __restrict__ xs, const float* __restrict__ xz,  /* z = xz cols [DI,2*DI) */
    const float* __restrict__ draw, const float* __restrict__ Bp, const float* __restrict__ Cp,
    const float* __restrict__ Aneg, const float* __restrict__ wdtv,
    const float* __restrict__ bdtv, const float* __restrict__ Dv,
    unsigned short* __restrict__ yg)
{
  extern __shared__ float smem[];
  float* sdr = smem;               /* 64    */
  float* sB  = sdr + 64;           /* 64*16 */
  float* sC  = sB  + 1024;         /* 64*16 */
  float* sxc = sC  + 1024;         /* 64*256 */
  float* szc = sxc + 16384;        /* 64*256 */

  const int tid   = threadIdx.x;
  const int b     = blockIdx.x >> 3;
  const int dBase = (blockIdx.x & 7) * 256;
  const int d     = dBase + tid;

  const float wdt = wdtv[d], bdt = bdtv[d], Dd = Dv[d];
  float Ar[DSTATE], h[DSTATE];
  #pragma unroll
  for (int n = 0; n < DSTATE; ++n) { Ar[n] = Aneg[(size_t)d*DSTATE + n]; h[n] = 0.f; }

  for (int l0 = 0; l0 < LL; l0 += 64) {
    const int tokBase = b*LL + l0;
    for (int i = tid; i < 64;   i += 256) sdr[i] = draw[tokBase + i];
    for (int i = tid; i < 1024; i += 256) {
      sB[i] = Bp[(size_t)tokBase*DSTATE + i];
      sC[i] = Cp[(size_t)tokBase*DSTATE + i];
    }
    for (int i = tid; i < 64*256; i += 256) {
      int t = i >> 8, dd = i & 255;
      sxc[i] = xs[(size_t)(tokBase + t)*DI  + dBase + dd];
      szc[i] = xz[(size_t)(tokBase + t)*NXZ + DI + dBase + dd];
    }
    __syncthreads();

    for (int t = 0; t < 64; ++t) {
      float dr = sdr[t];
      float xv = sxc[(t << 8) + tid];
      float zv = szc[(t << 8) + tid];
      float xa = fmaf(dr, wdt, bdt);
      float dt = (xa > 15.f) ? xa : __logf(1.f + __expf(xa));   /* softplus */
      float dx = dt * xv;
      const float* bt = sB + (t << 4);
      const float* ct = sC + (t << 4);
      float y = 0.f;
      #pragma unroll
      for (int n = 0; n < DSTATE; ++n) {
        float a = __expf(dt * Ar[n]);
        h[n] = fmaf(a, h[n], dx * bt[n]);
        y    = fmaf(h[n], ct[n], y);
      }
      float sig = 1.f / (1.f + __expf(-zv));
      float o   = (y + xv*Dd) * (zv * sig);
      yg[(size_t)(tokBase + t)*DI + d] = f2bf(o);
    }
    __syncthreads();
  }
}

/* ---------------- launch ---------------- */

extern "C" void kernel_launch(void* const* d_in, const int* in_sizes, int n_in,
                              void* d_out, int out_size, void* d_ws, size_t ws_size,
                              hipStream_t stream)
{
  const float* x     = (const float*)d_in[0];
  const float* W_in  = (const float*)d_in[1];
  const float* convw = (const float*)d_in[2];
  const float* convb = (const float*)d_in[3];
  const float* W_x   = (const float*)d_in[4];
  const float* w_dt  = (const float*)d_in[5];
  const float* b_dt  = (const float*)d_in[6];
  const float* A_log = (const float*)d_in[7];
  const float* Dvec  = (const float*)d_in[8];
  const float* W_out = (const float*)d_in[9];
  float* out = (float*)d_out;

  char* ws = (char*)d_ws;
  size_t off = 0;
  auto alloc = [&](size_t bytes) -> void* {
    void* p = ws + off;
    off = (off + bytes + 255) & ~(size_t)255;
    return p;
  };
  unsigned short* xbf   = (unsigned short*)alloc((size_t)MTOK*DM*2);     /* 16 MB */
  unsigned short* WinT  = (unsigned short*)alloc((size_t)NXZ*DM*2);      /*  8 MB */
  unsigned short* WoutT = (unsigned short*)alloc((size_t)DM*DI*2);       /*  4 MB */
  float*          Aneg  = (float*)alloc((size_t)DI*DSTATE*4);
  float*          WxT   = (float*)alloc((size_t)33*DI*4);
  float*          xzbuf = (float*)alloc((size_t)MTOK*NXZ*4);             /* 134 MB */
  float*          xsbuf = (float*)alloc((size_t)MTOK*DI*4);              /*  67 MB */
  float*          drawb = (float*)alloc((size_t)MTOK*4);
  float*          Bpb   = (float*)alloc((size_t)MTOK*DSTATE*4);
  float*          Cpb   = (float*)alloc((size_t)MTOK*DSTATE*4);
  unsigned short* ygb   = (unsigned short*)alloc((size_t)MTOK*DI*2);     /*  33 MB */

  /* prep: bf16 conversions / transposes, A = -exp(A_log) */
  int n1 = MTOK*DM;
  k_cvt_bf16      <<<(n1+255)/256,        256, 0, stream>>>(x, xbf, n1);
  k_transpose_bf16<<<((DM*NXZ)+255)/256,  256, 0, stream>>>(W_in,  WinT,  DM, NXZ);
  k_transpose_bf16<<<((DI*DM)+255)/256,   256, 0, stream>>>(W_out, WoutT, DI, DM);
  k_negexp        <<<((DI*DSTATE)+255)/256,256,0, stream>>>(A_log, Aneg, DI*DSTATE);
  k_transpose_f32 <<<((DI*33)+255)/256,   256, 0, stream>>>(W_x, WxT, DI, 33);

  const size_t gemmShmem = (size_t)4 * 128 * 64 * sizeof(unsigned short);  /* 64 KB */

  /* xz = x @ W_in  (TDM-staged WMMA, 68.7 GF) */
  dim3 g1(MTOK/128, NXZ/128);
  k_gemm_bf16<<<g1, 256, gemmShmem, stream>>>(xbf, WinT, xzbuf, MTOK, NXZ, DM);

  /* xs = silu(conv(xc)) */
  int nconv = MTOK*DI;
  k_conv_silu<<<(nconv+255)/256, 256, 0, stream>>>(xzbuf, convw, convb, xsbuf, nconv);

  /* x_dbl = xs @ W_x  ->  d_raw, B, C */
  k_xdbl<<<MTOK/4, 128, 0, stream>>>(xsbuf, WxT, drawb, Bpb, Cpb);

  /* selective scan + skip + gate -> bf16 */
  size_t shmem = (size_t)(64 + 1024 + 1024 + 16384 + 16384) * sizeof(float);
  k_scan_gate<<<BB*(DI/256), 256, shmem, stream>>>(xsbuf, xzbuf, drawb, Bpb, Cpb,
                                                   Aneg, w_dt, b_dt, Dvec, ygb);

  /* out = y @ W_out  (TDM-staged WMMA, 34.4 GF) */
  dim3 g2(MTOK/128, DM/128);
  k_gemm_bf16<<<g2, 256, gemmShmem, stream>>>(ygb, WoutT, out, MTOK, DM, DI);
}
